// GAT_77068893159934
// MI455X (gfx1250) — matrix operand order repaired and checked by reading, not compiled, not run on hardware
//
#include <hip/hip_runtime.h>
#include <cmath>

#define NEG_SLOPE 0.2f
#define EPS_F 1e-16f

typedef __attribute__((ext_vector_type(16))) __bf16 v16bf;
typedef __attribute__((ext_vector_type(8)))  __bf16 v8bf;
typedef __attribute__((ext_vector_type(2)))  __bf16 v2bf;
typedef __attribute__((ext_vector_type(8)))  float  v8f;

// ---------------- utility ----------------

__global__ void fill_f32(float* __restrict__ p, float v, long long n) {
    long long i = (long long)blockIdx.x * blockDim.x + threadIdx.x;
    if (i < n) p[i] = v;
}

__device__ inline void atomicMaxF(float* addr, float v) {
    // bit-pattern trick: works with addr initialized to -inf
    if (v >= 0.0f) atomicMax((int*)addr, __float_as_int(v));
    else           atomicMin((unsigned int*)addr, __float_as_uint(v));
}

__device__ inline v16bf cat16(v8bf lo, v8bf hi) {
    return __builtin_shufflevector(lo, hi, 0, 1, 2, 3, 4, 5, 6, 7,
                                           8, 9, 10, 11, 12, 13, 14, 15);
}

// ---------------- GEMM 1: h1[N,64] = x[N,512] @ W1[512,64] (bf16 WMMA, f32 acc) ----------------
// LDS layout: rows padded to 40 bf16 (80B, 16B-aligned) to spread banks.
// A stored row-major [row][k]; B stored TRANSPOSED [col][k] so each lane's
// fragment is two contiguous 16B runs -> ds_load_b128 pairs.

__global__ __launch_bounds__(128) void gemm1_wmma(const float* __restrict__ x,
                                                  const float* __restrict__ W1,
                                                  float* __restrict__ h1, int Nn) {
    __shared__ __align__(16) __bf16 As[64][40];
    __shared__ __align__(16) __bf16 BsT[64][40];
    const int tid  = threadIdx.x;
    const int wave = tid >> 5;
    const int lane = tid & 31;
    const int row0 = blockIdx.x * 64;

    v8f acc[4] = {{}, {}, {}, {}};

    for (int k0 = 0; k0 < 512; k0 += 32) {
        // stage A tile (64 rows x 32 k) as packed bf16 pairs
        for (int i = tid; i < 64 * 16; i += 128) {
            int r = i >> 4, cp = (i & 15) << 1;
            int g = row0 + r; if (g >= Nn) g = Nn - 1;
            const float* px = x + (size_t)g * 512 + k0 + cp;
            v2bf pk; pk[0] = (__bf16)px[0]; pk[1] = (__bf16)px[1];
            *(v2bf*)&As[r][cp] = pk;
        }
        // stage B tile transposed: BsT[col][k]; consecutive tid -> consecutive col (coalesced)
        for (int i = tid; i < 64 * 16; i += 128) {
            int c = i & 63, kk = (i >> 6) << 1;
            v2bf pk;
            pk[0] = (__bf16)W1[(size_t)(k0 + kk) * 64 + c];
            pk[1] = (__bf16)W1[(size_t)(k0 + kk + 1) * 64 + c];
            *(v2bf*)&BsT[c][kk] = pk;
        }
        __syncthreads();

        // A fragment: lane L -> M = L&15; elems 0..15 = K khalf..khalf+15
        const int m     = wave * 16 + (lane & 15);
        const int khalf = (lane >> 4) * 16;
        v16bf a = cat16(*(const v8bf*)&As[m][khalf],
                        *(const v8bf*)&As[m][khalf + 8]);

        // B fragment: lane L -> Ncol = L&15; K run = kb..kb+15 (contiguous in BsT)
        const int kb = (lane >> 4) * 16;
        const int nl = lane & 15;
        #pragma unroll
        for (int cb = 0; cb < 4; ++cb) {
            const __bf16* brow = &BsT[cb * 16 + nl][kb];
            v16bf b = cat16(*(const v8bf*)brow, *(const v8bf*)(brow + 8));
            acc[cb] = __builtin_amdgcn_wmma_f32_16x16x32_bf16(false, a, false, b,
                                                              (short)0, acc[cb], false, false);
        }
        __syncthreads();
    }

    // C/D layout: VGPR r -> M = r + 8*(lane>=16), Ncol = lane&15
    const int nl    = lane & 15;
    const int mBase = row0 + wave * 16 + ((lane >> 4) ? 8 : 0);
    #pragma unroll
    for (int r = 0; r < 8; ++r) {
        int g = mBase + r;
        if (g < Nn) {
            float* dst = h1 + (size_t)g * 64 + nl;
            dst[0]  = acc[0][r];
            dst[16] = acc[1][r];
            dst[32] = acc[2][r];
            dst[48] = acc[3][r];
        }
    }
}

// ---------------- GEMM 2: h2[N,40] = hact[N,64] @ W2[64,40] ----------------

__global__ __launch_bounds__(128) void gemm2_wmma(const float* __restrict__ hact,
                                                  const float* __restrict__ W2,
                                                  float* __restrict__ h2, int Nn) {
    __shared__ __align__(16) __bf16 As[64][40];
    __shared__ __align__(16) __bf16 BsT[48][40];   // 40 cols padded to 48
    const int tid  = threadIdx.x;
    const int wave = tid >> 5;
    const int lane = tid & 31;
    const int row0 = blockIdx.x * 64;

    v8f acc[3] = {{}, {}, {}};

    for (int k0 = 0; k0 < 64; k0 += 32) {
        for (int i = tid; i < 64 * 16; i += 128) {
            int r = i >> 4, cp = (i & 15) << 1;
            int g = row0 + r; if (g >= Nn) g = Nn - 1;
            const float* ph = hact + (size_t)g * 64 + k0 + cp;
            v2bf pk; pk[0] = (__bf16)ph[0]; pk[1] = (__bf16)ph[1];
            *(v2bf*)&As[r][cp] = pk;
        }
        for (int i = tid; i < 48 * 16; i += 128) {
            int c = i % 48, kk = (i / 48) << 1;
            v2bf pk;
            pk[0] = (__bf16)((c < 40) ? W2[(size_t)(k0 + kk) * 40 + c] : 0.0f);
            pk[1] = (__bf16)((c < 40) ? W2[(size_t)(k0 + kk + 1) * 40 + c] : 0.0f);
            *(v2bf*)&BsT[c][kk] = pk;
        }
        __syncthreads();

        const int m     = wave * 16 + (lane & 15);
        const int khalf = (lane >> 4) * 16;
        v16bf a = cat16(*(const v8bf*)&As[m][khalf],
                        *(const v8bf*)&As[m][khalf + 8]);

        const int kb = (lane >> 4) * 16;
        const int nl = lane & 15;
        #pragma unroll
        for (int cb = 0; cb < 3; ++cb) {
            const __bf16* brow = &BsT[cb * 16 + nl][kb];
            v16bf b = cat16(*(const v8bf*)brow, *(const v8bf*)(brow + 8));
            acc[cb] = __builtin_amdgcn_wmma_f32_16x16x32_bf16(false, a, false, b,
                                                              (short)0, acc[cb], false, false);
        }
        __syncthreads();
    }

    const int nl    = lane & 15;
    const int mBase = row0 + wave * 16 + ((lane >> 4) ? 8 : 0);
    #pragma unroll
    for (int r = 0; r < 8; ++r) {
        int g = mBase + r;
        if (g < Nn) {
            if (nl      < 40) h2[(size_t)g * 40 + nl]      = acc[0][r];
            if (16 + nl < 40) h2[(size_t)g * 40 + 16 + nl] = acc[1][r];
            if (32 + nl < 40) h2[(size_t)g * 40 + 32 + nl] = acc[2][r];
        }
    }
}

// ---------------- attention logits per node ----------------
// a_src[n*H+h] = sum_d h[n,H,D] * att_src[h,d]; same for dst

__global__ void node_logits(const float* __restrict__ h,
                            const float* __restrict__ att_s,
                            const float* __restrict__ att_d,
                            float* __restrict__ a_s, float* __restrict__ a_d,
                            int Nn, int H, int D) {
    int idx = blockIdx.x * blockDim.x + threadIdx.x;
    if (idx >= Nn * H) return;
    int n = idx / H, hh = idx % H;
    const float* row = h + (size_t)n * H * D + (size_t)hh * D;
    float ss = 0.0f, dd = 0.0f;
    for (int d = 0; d < D; ++d) {
        float v = row[d];
        ss += v * att_s[hh * D + d];
        dd += v * att_d[hh * D + d];
    }
    a_s[idx] = ss;
    a_d[idx] = dd;
}

// ---------------- edge pass 1: leaky relu logit + segment max ----------------

__global__ void edge_leaky_max(const int* __restrict__ srcA, const int* __restrict__ dstA,
                               const float* __restrict__ a_s, const float* __restrict__ a_d,
                               float* __restrict__ e, float* __restrict__ m,
                               int E, long long Etot, int H) {
    long long idx = (long long)blockIdx.x * blockDim.x + threadIdx.x;
    if (idx >= Etot * H) return;
    long long eid = idx / H;
    int hh = (int)(idx % H);
    int s_, d_;
    if (eid < E) { s_ = srcA[eid]; d_ = dstA[eid]; }
    else         { s_ = d_ = (int)(eid - E); }
    float v = a_s[(size_t)s_ * H + hh] + a_d[(size_t)d_ * H + hh];
    v = (v > 0.0f) ? v : NEG_SLOPE * v;
    e[idx] = v;
    atomicMaxF(&m[(size_t)d_ * H + hh], v);
}

// ---------------- edge pass 2: exp(e - m[dst]) + segment sum ----------------

__global__ void edge_exp_sum(const int* __restrict__ dstA,
                             float* __restrict__ e, const float* __restrict__ m,
                             float* __restrict__ s,
                             int E, long long Etot, int H) {
    long long idx = (long long)blockIdx.x * blockDim.x + threadIdx.x;
    if (idx >= Etot * H) return;
    long long eid = idx / H;
    int hh = (int)(idx % H);
    int d_ = (eid < E) ? dstA[eid] : (int)(eid - E);
    float t = __expf(e[idx] - m[(size_t)d_ * H + hh]);
    e[idx] = t;
    atomicAdd(&s[(size_t)d_ * H + hh], t);
}

// ---------------- edge pass 3: alpha-weighted scatter-add ----------------

__global__ void edge_aggregate(const int* __restrict__ srcA, const int* __restrict__ dstA,
                               const float* __restrict__ e, const float* __restrict__ s,
                               const float* __restrict__ h, float* __restrict__ out,
                               int E, long long Etot, int H, int D) {
    const int F = H * D;
    long long idx = (long long)blockIdx.x * blockDim.x + threadIdx.x;
    if (idx >= Etot * F) return;
    long long eid = idx / F;
    int f = (int)(idx % F);
    int hh = f / D;
    int s_, d_;
    if (eid < E) { s_ = srcA[eid]; d_ = dstA[eid]; }
    else         { s_ = d_ = (int)(eid - E); }
    float alpha = e[(size_t)eid * H + hh] / (s[(size_t)d_ * H + hh] + EPS_F);
    atomicAdd(&out[(size_t)d_ * F + f], alpha * h[(size_t)s_ * F + f]);
}

// ---------------- bias + ELU (layer 1 output, in place) ----------------

__global__ void bias_elu(float* __restrict__ p, const float* __restrict__ bias, long long n, int F) {
    long long i = (long long)blockIdx.x * blockDim.x + threadIdx.x;
    if (i >= n) return;
    float v = p[i] + bias[i % F];
    p[i] = (v > 0.0f) ? v : (__expf(v) - 1.0f);
}

// ---------------- final: + b2, log_softmax over 40 classes ----------------

__global__ void final_logsoftmax(const float* __restrict__ out2, const float* __restrict__ b2,
                                 float* __restrict__ y, int Nn) {
    int n = blockIdx.x * blockDim.x + threadIdx.x;
    if (n >= Nn) return;
    float row[40];
    float mx = -3.402823466e38f;
    #pragma unroll
    for (int c = 0; c < 40; ++c) {
        float v = out2[(size_t)n * 40 + c] + b2[c];
        row[c] = v;
        mx = fmaxf(mx, v);
    }
    float sum = 0.0f;
    #pragma unroll
    for (int c = 0; c < 40; ++c) sum += __expf(row[c] - mx);
    float l = mx + __logf(sum);
    #pragma unroll
    for (int c = 0; c < 40; ++c) y[(size_t)n * 40 + c] = row[c] - l;
}

// ---------------- launch ----------------

static inline int blocks_for(long long n, int bs) { return (int)((n + bs - 1) / bs); }

extern "C" void kernel_launch(void* const* d_in, const int* in_sizes, int n_in,
                              void* d_out, int out_size, void* d_ws, size_t ws_size,
                              hipStream_t stream) {
    const float* x        = (const float*)d_in[0];
    const int*   eidx     = (const int*)d_in[1];
    const float* W1       = (const float*)d_in[2];
    const float* att_src1 = (const float*)d_in[3];
    const float* att_dst1 = (const float*)d_in[4];
    const float* b1       = (const float*)d_in[5];
    const float* W2       = (const float*)d_in[6];
    const float* att_src2 = (const float*)d_in[7];
    const float* att_dst2 = (const float*)d_in[8];
    const float* b2       = (const float*)d_in[9];
    float*       y        = (float*)d_out;

    const int Nn = in_sizes[0] / 512;         // 100000
    const int E  = in_sizes[1] / 2;           // 1600000
    const long long Etot = (long long)E + Nn; // with self-loops
    const int* srcA = eidx;
    const int* dstA = eidx + E;

    float* ws = (float*)d_ws;
    const size_t N64 = (size_t)Nn * 64, N8 = (size_t)Nn * 8, N40 = (size_t)Nn * 40;
    size_t off = 0;
    float* h1    = ws + off; off += N64;
    float* asrc1 = ws + off; off += N8;
    float* adst1 = ws + off; off += N8;
    float* m1    = ws + off; off += N8;
    float* s1    = ws + off; off += N8;
    float* out1  = ws + off; off += N64;      // becomes ELU(h1_agg), input to layer 2
    float* e1    = ws + off;                  // Etot*8 floats
    // layer-2 buffers alias the e1 region (e1 is dead once layer-1 aggregation finishes)
    float* l2    = e1; size_t o2 = 0;
    float* h2    = l2 + o2; o2 += N40;
    float* asrc2 = l2 + o2; o2 += Nn;
    float* adst2 = l2 + o2; o2 += Nn;
    float* m2    = l2 + o2; o2 += Nn;
    float* s2    = l2 + o2; o2 += Nn;
    float* e2    = l2 + o2; o2 += Etot;
    float* out2  = l2 + o2; o2 += N40;

    const int BS = 256;
    const float NEG_INF = -INFINITY;

    // ---- layer 1 ----
    fill_f32<<<blocks_for(N8, BS), BS, 0, stream>>>(m1, NEG_INF, (long long)N8);
    fill_f32<<<blocks_for(N8, BS), BS, 0, stream>>>(s1, 0.0f, (long long)N8);
    fill_f32<<<blocks_for(N64, BS), BS, 0, stream>>>(out1, 0.0f, (long long)N64);

    gemm1_wmma<<<(Nn + 63) / 64, 128, 0, stream>>>(x, W1, h1, Nn);
    node_logits<<<blocks_for((long long)Nn * 8, BS), BS, 0, stream>>>(h1, att_src1, att_dst1,
                                                                      asrc1, adst1, Nn, 8, 8);
    edge_leaky_max<<<blocks_for(Etot * 8, BS), BS, 0, stream>>>(srcA, dstA, asrc1, adst1,
                                                                e1, m1, E, Etot, 8);
    edge_exp_sum<<<blocks_for(Etot * 8, BS), BS, 0, stream>>>(dstA, e1, m1, s1, E, Etot, 8);
    edge_aggregate<<<blocks_for(Etot * 64, BS), BS, 0, stream>>>(srcA, dstA, e1, s1, h1, out1,
                                                                 E, Etot, 8, 8);
    bias_elu<<<blocks_for(N64, BS), BS, 0, stream>>>(out1, b1, (long long)N64, 64);

    // ---- layer 2 ----
    fill_f32<<<blocks_for(Nn, BS), BS, 0, stream>>>(m2, NEG_INF, (long long)Nn);
    fill_f32<<<blocks_for(Nn, BS), BS, 0, stream>>>(s2, 0.0f, (long long)Nn);
    fill_f32<<<blocks_for(N40, BS), BS, 0, stream>>>(out2, 0.0f, (long long)N40);

    gemm2_wmma<<<(Nn + 63) / 64, 128, 0, stream>>>(out1, W2, h2, Nn);
    node_logits<<<blocks_for((long long)Nn, BS), BS, 0, stream>>>(h2, att_src2, att_dst2,
                                                                  asrc2, adst2, Nn, 1, 40);
    edge_leaky_max<<<blocks_for(Etot, BS), BS, 0, stream>>>(srcA, dstA, asrc2, adst2,
                                                            e2, m2, E, Etot, 1);
    edge_exp_sum<<<blocks_for(Etot, BS), BS, 0, stream>>>(dstA, e2, m2, s2, E, Etot, 1);
    edge_aggregate<<<blocks_for(Etot * 40, BS), BS, 0, stream>>>(srcA, dstA, e2, s2, h2, out2,
                                                                 E, Etot, 1, 40);
    final_logsoftmax<<<blocks_for(Nn, BS), BS, 0, stream>>>(out2, b2, y, Nn);
}
